// LookupFFN_67018669687167
// MI455X (gfx1250) — matrix operand order repaired and checked by reading, not compiled, not run on hardware
//
#include <hip/hip_runtime.h>

// Problem constants: B=4, S=4096, D_IN=1024, D_OUT=1024, N_CENT=1024
#define NROWS   16384       // B*S
#define KDIM    1024
#define NCENT   1024
#define DOUT    1024
#define WG_ROWS 64          // rows per workgroup (4 waves x 16)
#define THREADS 128
#define NTILES  64          // NCENT / 16
#define XS_STRIDE 1032      // padded row stride (halves): conflict-minimal ds_load_b128
#define BS_STRIDE 1032

// dynamic LDS layout (bytes)
#define XS_BYTES  (WG_ROWS * XS_STRIDE * 2)   // 132096
#define BS_OFF    XS_BYTES
#define BS_BYTES  (16 * BS_STRIDE * 2)        // 33024
#define FIDX_OFF  (BS_OFF + BS_BYTES)         // 165120
#define LDS_TOTAL (FIDX_OFF + WG_ROWS * 4)    // 165376

typedef __attribute__((ext_vector_type(16))) _Float16 v16h;
typedef __attribute__((ext_vector_type(8)))  _Float16 v8h;
typedef __attribute__((ext_vector_type(8)))  float    v8f;

// ---------------------------------------------------------------------------
// Kernel 1: centroids f32 -> f16 (row-major) + exact fp32 squared norms
// ---------------------------------------------------------------------------
__global__ __launch_bounds__(256)
void lookupffn_prep_kernel(const float* __restrict__ cent,
                           _Float16* __restrict__ cf16,
                           float* __restrict__ csq)
{
    __shared__ float red[256];
    const int row = blockIdx.x;
    const int tid = threadIdx.x;

    const float4* src = (const float4*)(cent + (long)row * KDIM);
    float4 v = src[tid];
    float acc = v.x * v.x + v.y * v.y + v.z * v.z + v.w * v.w;

    _Float16* dst = cf16 + (long)row * KDIM + tid * 4;
    dst[0] = (_Float16)v.x;
    dst[1] = (_Float16)v.y;
    dst[2] = (_Float16)v.z;
    dst[3] = (_Float16)v.w;

    red[tid] = acc;
    __syncthreads();
    for (int s = 128; s > 0; s >>= 1) {
        if (tid < s) red[tid] += red[tid + s];
        __syncthreads();
    }
    if (tid == 0) csq[row] = red[0];
}

// ---------------------------------------------------------------------------
// Kernel 2: fused WMMA distance-GEMM + argmin + gather.
//  - 4 waves, each owning 16 of the WG's 64 rows (x staged once as f16 in LDS)
//  - all waves sweep the same 64 centroid tiles; each 32KB B tile is staged in
//    LDS via async global->LDS (ASYNCcnt) and shared by all 4 waves
//  - per tile: 32x v_wmma_f32_16x16x32_f16 over K=1024 into one v8f acc
//  - score = c_sq[n] - 2*dot  (x^2 is row-constant -> irrelevant to argmin)
// ---------------------------------------------------------------------------
__global__ __launch_bounds__(THREADS)
void lookupffn_main_kernel(const float* __restrict__ x,
                           const _Float16* __restrict__ cf16,
                           const float* __restrict__ csq,
                           const float* __restrict__ table,
                           const float* __restrict__ bias,
                           float* __restrict__ out)
{
    extern __shared__ char lds_raw[];
    _Float16* xs  = (_Float16*)lds_raw;               // [64][1032] f16
    _Float16* bs  = (_Float16*)(lds_raw + BS_OFF);    // [16][1032] f16 (B tile)
    int*      fidx = (int*)(lds_raw + FIDX_OFF);      // [64]

    const int tid  = threadIdx.x;
    const int wave = tid >> 5;
    const int lane = tid & 31;
    const long rowBase = (long)blockIdx.x * WG_ROWS;

    // ---- Phase 1: stage x tile [64 x 1024] f32 -> f16 into LDS ----
    {
        const float4* xv = (const float4*)(x + rowBase * KDIM);
        for (int i = tid; i < WG_ROWS * KDIM / 4; i += THREADS) {
            int r  = i >> 8;          // 256 float4 per row
            int k4 = i & 255;
            float4 v = xv[i];
            _Float16* dst = xs + r * XS_STRIDE + k4 * 4;
            dst[0] = (_Float16)v.x;
            dst[1] = (_Float16)v.y;
            dst[2] = (_Float16)v.z;
            dst[3] = (_Float16)v.w;
        }
    }

    const int mrow = lane & 15;
    const int half = lane >> 4;

    float minv[8];
    int   mini[8];
#pragma unroll
    for (int v = 0; v < 8; ++v) { minv[v] = 3.402823e38f; mini[v] = 0; }

    // A fragment base (16-bit A 16x32 layout): this wave's row wave*16 + mrow
    //   VGPR0-3: K = 8*half + 0..7 ; VGPR4-7: K = 16 + 8*half + 0..7
    const _Float16* arow = xs + (wave * 16 + mrow) * XS_STRIDE + 8 * half;
    // B fragment base (16-bit B 32x16 layout): lane holds column mrow of tile,
//   K = 8*half + 0..7 and 16 + 8*half + 0..7 (matches A)
    const _Float16* browl = bs + mrow * BS_STRIDE + 8 * half;

    for (int nt = 0; nt < NTILES; ++nt) {
        // wait until previous tile's consumers are done (and, at nt==0, xs staged)
        __syncthreads();

        // ---- stage B tile [16 x 1024] f16 via async global->LDS copies ----
        {
            const _Float16* gbase = cf16 + (size_t)nt * 16 * KDIM;
#pragma unroll
            for (int j = 0; j < 16; ++j) {
                int c  = tid + j * THREADS;     // b128 chunk id: 2048 per tile
                int r  = c >> 7;                // 128 chunks per row
                int k8 = (c & 127) << 3;
                unsigned lds_off = (unsigned)(uintptr_t)(bs + r * BS_STRIDE + k8);
                unsigned long long gaddr =
                    (unsigned long long)(uintptr_t)(gbase + (size_t)r * KDIM + k8);
                asm volatile("global_load_async_to_lds_b128 %0, %1, off"
                             :: "v"(lds_off), "v"(gaddr) : "memory");
            }
            asm volatile("s_wait_asynccnt 0x0" ::: "memory");
        }
        __syncthreads();

        // ---- compute: 16x16 tile, K = 1024 ----
        const int nc = nt * 16 + mrow;          // this lane's centroid column
        const float cs = csq[nc];

        v8f acc = {};
#pragma unroll 4
        for (int kk = 0; kk < 32; ++kk) {
            v16h a, b;
            ((v8h*)&a)[0] = *(const v8h*)(arow + 32 * kk);
            ((v8h*)&a)[1] = *(const v8h*)(arow + 32 * kk + 16);
            ((v8h*)&b)[0] = *(const v8h*)(browl + 32 * kk);
            ((v8h*)&b)[1] = *(const v8h*)(browl + 32 * kk + 16);
            acc = __builtin_amdgcn_wmma_f32_16x16x32_f16(
                false, a, false, b, (short)0, acc, false, false);
        }

#pragma unroll
        for (int v = 0; v < 8; ++v) {
            float s = cs - 2.0f * acc[v];
            if (s < minv[v]) { minv[v] = s; mini[v] = nc; }   // nc increases with nt
        }
    }

    // ---- lane argmin reduction within each 16-lane half ----
    // slot v of lane holds row (wave*16 + v + 8*half); rows owned by one wave only
#pragma unroll
    for (int v = 0; v < 8; ++v) {
        float mv = minv[v];
        int   mi = mini[v];
        for (int off = 1; off < 16; off <<= 1) {
            float ov = __shfl_xor(mv, off, 32);
            int   oi = __shfl_xor(mi, off, 32);
            if (ov < mv || (ov == mv && oi < mi)) { mv = ov; mi = oi; }
        }
        minv[v] = mv;
        mini[v] = mi;
    }
    if (lane == 0 || lane == 16) {
        const int base = wave * 16 + half * 8;
#pragma unroll
        for (int v = 0; v < 8; ++v) fidx[base + v] = mini[v];
    }
    __syncthreads();

    // ---- gather table[idx] + bias -> out (64 rows) ----
    for (int i = tid; i < WG_ROWS * DOUT / 4; i += THREADS) {
        int r  = i >> 8;
        int k4 = i & 255;
        const float4* trow = (const float4*)(table + (long)fidx[r] * DOUT);
        float4 t  = trow[k4];
        float4 bz = ((const float4*)bias)[k4];
        float4 o;
        o.x = t.x + bz.x; o.y = t.y + bz.y; o.z = t.z + bz.z; o.w = t.w + bz.w;
        ((float4*)(out + (rowBase + r) * DOUT))[k4] = o;
    }
}

// ---------------------------------------------------------------------------
// inputs: x [4,4096,1024] f32, input_centroids [1024,1024] f32,
//         lookup_table_fc2 [1024,1024] f32, fc2_bias [1024] f32
// output: [4,4096,1024] f32
// ws: [0,2MB) centroids f16, [2MB,2MB+4KB) c_sq f32
// ---------------------------------------------------------------------------
extern "C" void kernel_launch(void* const* d_in, const int* in_sizes, int n_in,
                              void* d_out, int out_size, void* d_ws, size_t ws_size,
                              hipStream_t stream)
{
    const float* x     = (const float*)d_in[0];
    const float* cent  = (const float*)d_in[1];
    const float* table = (const float*)d_in[2];
    const float* bias  = (const float*)d_in[3];
    float* out = (float*)d_out;

    _Float16* cf16 = (_Float16*)d_ws;
    float*    csq  = (float*)((char*)d_ws + (size_t)NCENT * KDIM * sizeof(_Float16));

    // allow >64KB dynamic LDS (CDNA5 WGP has 320KB); ignore error if unsupported
    (void)hipFuncSetAttribute((const void*)lookupffn_main_kernel,
                              hipFuncAttributeMaxDynamicSharedMemorySize, LDS_TOTAL);

    lookupffn_prep_kernel<<<NCENT, 256, 0, stream>>>(cent, cf16, csq);
    lookupffn_main_kernel<<<NROWS / WG_ROWS, THREADS, LDS_TOTAL, stream>>>(
        x, cf16, csq, table, bias, out);
}
